// RSSM_43946105373236
// MI455X (gfx1250) — compile-verified, matrix operand order checked by reading, not run on hardware
//
#include <hip/hip_runtime.h>

// ---------------------------------------------------------------------------
// RSSM scan for MI455X (gfx1250): bf16 WMMA GEMMs + fp32 state/gates.
// B=512 T=128 S=64 D=1024 H=1024 E=1024 A=16
// GEMM tiles staged with CDNA5 async global->LDS loads (ASYNCcnt).
// ---------------------------------------------------------------------------

typedef unsigned short u16;
typedef unsigned int   u32;

typedef __bf16 bf16;
typedef __attribute__((ext_vector_type(16))) __bf16 v16bf;
typedef __attribute__((ext_vector_type(8)))  __bf16 v8bf;
typedef __attribute__((ext_vector_type(8)))  float  v8f;

#define BB 512
#define TT 128
#define SS 64
#define DD 1024
#define HH 1024
#define EE 1024
#define AA 16
#define KAS 96   // 80 padded to multiple of 32

__device__ __forceinline__ u16 f2bf(float f) {
  union { float f; u32 u; } c; c.f = f;
  u32 u = c.u;
  u += 0x7FFFu + ((u >> 16) & 1u);   // round-to-nearest-even-ish
  return (u16)(u >> 16);
}

__device__ __forceinline__ float softplus_f(float x) {
  return x > 0.0f ? x + log1pf(__expf(-x)) : log1pf(__expf(x));
}

// CDNA5 async global->LDS 16B copy (per lane), tracked by ASYNCcnt.
__device__ __forceinline__ void async_b128(u32 lds_off, const u16* g) {
  asm volatile("global_load_async_to_lds_b128 %0, %1, off"
               :: "v"(lds_off), "v"(g) : "memory");
}
__device__ __forceinline__ void wait_async0() {
  asm volatile("s_wait_asynccnt 0x0" ::: "memory");
}

// --------------------------- weight conversion ------------------------------
__global__ void convert_pad_bf16(const float* __restrict__ src, u16* __restrict__ dst,
                                 int N, int K, int Kpad) {
  int idx = blockIdx.x * blockDim.x + threadIdx.x;
  if (idx >= N * Kpad) return;
  int n = idx / Kpad, k = idx - n * Kpad;
  float v = (k < K) ? src[(size_t)n * K + k] : 0.0f;
  dst[idx] = f2bf(v);
}

__global__ void zero_f32(float* __restrict__ p, int n) {
  int i = blockIdx.x * blockDim.x + threadIdx.x;
  if (i < n) p[i] = 0.0f;
}

// ------------------------------- WMMA GEMM ----------------------------------
// C[512, N] = act( A[512, K]·Wᵀ + bias ), A bf16 row-major (lda), W bf16 [N,K]
// row-major (ldw).  flags bit0 = ELU, bit1 = store bf16 (else fp32).
// Workgroup: 256 thr = 8 waves; tile 128x128; wave tile 64x32 = 4x2 WMMAs.
// K tiles double-buffered in LDS, filled by async global->LDS loads.
__global__ __launch_bounds__(256)
void wmma_gemm(const u16* __restrict__ Abf, int lda,
               const u16* __restrict__ Wbf, int ldw,
               const float* __restrict__ bias,
               void* __restrict__ Cout, int N, int K, int flags) {
  __shared__ __align__(16) u16 lA[2][128][40];   // +8 pad, rows stay 16B aligned
  __shared__ __align__(16) u16 lW[2][128][40];

  const int tid  = threadIdx.x;
  const int lane = tid & 31;
  const int wave = tid >> 5;
  const int wm   = wave >> 2;          // 0..1  (M direction)
  const int wn   = wave & 3;           // 0..3  (N direction)
  const int lrow = lane & 15;
  const int ka   = (lane >> 4) << 3;   // 0 or 8 : K striping of 16-bit WMMA frags

  const int mBase = blockIdx.x * 128;
  const int nBase = blockIdx.y * 128;

  const int srow = tid >> 1;           // staging row 0..127
  const int sh   = tid & 1;            // staging half (16 bf16 each)

  const int nk = K >> 5;

  // accumulators init from bias (bias depends on n only)
  v8f acc[4][2];
#pragma unroll
  for (int nt = 0; nt < 2; ++nt) {
    float bv = bias[nBase + wn * 32 + nt * 16 + lrow];
#pragma unroll
    for (int mt = 0; mt < 4; ++mt)
#pragma unroll
      for (int e = 0; e < 8; ++e) acc[mt][nt][e] = bv;
  }

  // stage one 128x32 A tile + 128x32 W tile into LDS buffer `buf`
  // (4 async B128 per thread = this thread's 64B share)
  auto stage = [&](int buf, int kt) {
    const int k0 = kt << 5;
    const u16* pa = Abf + (size_t)(mBase + srow) * lda + k0 + sh * 16;
    const u16* pw = Wbf + (size_t)(nBase + srow) * ldw + k0 + sh * 16;
    u32 la = (u32)(uintptr_t)&lA[buf][srow][sh * 16];
    u32 lw = (u32)(uintptr_t)&lW[buf][srow][sh * 16];
    async_b128(la,      pa);
    async_b128(la + 16, pa + 8);
    async_b128(lw,      pw);
    async_b128(lw + 16, pw + 8);
  };

  stage(0, 0);
  wait_async0();
  __syncthreads();

  for (int kt = 0; kt < nk; ++kt) {
    const int  cur  = kt & 1, nxt = cur ^ 1;
    const bool more = (kt + 1 < nk);

    if (more) {
      stage(nxt, kt + 1);                 // async fill of other buffer
      if (kt + 2 < nk)                    // pull weight tile 2 ahead into L2
        __builtin_prefetch(Wbf + (size_t)(nBase + srow) * ldw + ((kt + 2) << 5), 0, 1);
    }

    // ---- load fragments from LDS (documented 16-bit WMMA K striping) ----
    v16bf af[4], bfr[2];
#pragma unroll
    for (int mt = 0; mt < 4; ++mt) {
      const bf16* rowp = (const bf16*)&lA[cur][wm * 64 + mt * 16 + lrow][0];
      v8bf lo = *(const v8bf*)(rowp + ka);
      v8bf hi = *(const v8bf*)(rowp + 16 + ka);
#pragma unroll
      for (int e = 0; e < 8; ++e) { af[mt][e] = lo[e]; af[mt][e + 8] = hi[e]; }
    }
#pragma unroll
    for (int nt = 0; nt < 2; ++nt) {
      const bf16* rowp = (const bf16*)&lW[cur][wn * 32 + nt * 16 + lrow][0];
      v8bf lo = *(const v8bf*)(rowp + ka);
      v8bf hi = *(const v8bf*)(rowp + 16 + ka);
#pragma unroll
      for (int e = 0; e < 8; ++e) { bfr[nt][e] = lo[e]; bfr[nt][e + 8] = hi[e]; }
    }

    // ---- 8 WMMAs: D = A x B + C, fp32 accumulate ----
#pragma unroll
    for (int mt = 0; mt < 4; ++mt)
#pragma unroll
      for (int nt = 0; nt < 2; ++nt)
        acc[mt][nt] = __builtin_amdgcn_wmma_f32_16x16x32_bf16(
            false, af[mt], false, bfr[nt], (short)0, acc[mt][nt], false, false);

    if (more) wait_async0();             // own async loads for `nxt` done
    __syncthreads();                     // ... and everyone else's too
  }

  // ---- epilogue: fused ELU + fp32/bf16 store ----
  const bool elu = (flags & 1) != 0;
  const bool obf = (flags & 2) != 0;
#pragma unroll
  for (int mt = 0; mt < 4; ++mt) {
    const int m0 = mBase + wm * 64 + mt * 16 + ((lane >> 4) << 3);
#pragma unroll
    for (int nt = 0; nt < 2; ++nt) {
      const int n = nBase + wn * 32 + nt * 16 + lrow;
#pragma unroll
      for (int r = 0; r < 8; ++r) {
        float v = acc[mt][nt][r];
        if (elu) v = v > 0.0f ? v : (__expf(v) - 1.0f);
        const size_t off = (size_t)(m0 + r) * N + n;
        if (obf) ((u16*)Cout)[off]  = f2bf(v);
        else     ((float*)Cout)[off] = v;
      }
    }
  }
}

// ----------------------------- per-step kernels -----------------------------
// mask state with (1-is_first), build bf16 [a|stoch|0pad] and post_in[:,1024:]
__global__ void step_prep(const float* __restrict__ actions,
                          const float* __restrict__ emb,
                          const int*   __restrict__ isf,
                          float* __restrict__ stoch_f, float* __restrict__ deter_f,
                          u16* __restrict__ deter_bf, u16* __restrict__ x_as,
                          u16* __restrict__ post_in, int t) {
  int idx = blockIdx.x * blockDim.x + threadIdx.x;   // B*D
  if (idx >= BB * DD) return;
  int b = idx >> 10, d = idx & 1023;
  float m = 1.0f - (float)isf[b * TT + t];
  float dv = deter_f[idx] * m;
  deter_f[idx]  = dv;
  deter_bf[idx] = f2bf(dv);
  post_in[(size_t)b * 2048 + 1024 + d] = f2bf(emb[((size_t)b * TT + t) * EE + d]);
  if (d < SS) {
    float sv = stoch_f[b * SS + d] * m;
    stoch_f[b * SS + d] = sv;
    x_as[b * KAS + AA + d] = f2bf(sv);
  }
  if (d < AA)              x_as[b * KAS + d] = f2bf(actions[((size_t)b * TT + t) * AA + d]);
  if (d >= 80 && d < KAS)  x_as[b * KAS + d] = 0;   // zero pad K 80..95
}

// GRU gate fusion: gi,gh [B,3D] -> new deter (fp32 + bf16 + post_in[:, :D] + out)
__global__ void gru_gate(const float* __restrict__ gi, const float* __restrict__ gh,
                         float* __restrict__ deter_f, u16* __restrict__ deter_bf,
                         u16* __restrict__ post_in, float* __restrict__ out_deter, int t) {
  int idx = blockIdx.x * blockDim.x + threadIdx.x;   // B*D
  if (idx >= BB * DD) return;
  int b = idx >> 10, d = idx & 1023;
  const float* gib = gi + (size_t)b * 3 * DD;
  const float* ghb = gh + (size_t)b * 3 * DD;
  float r = 1.0f / (1.0f + __expf(-(gib[d] + ghb[d])));
  float z = 1.0f / (1.0f + __expf(-(gib[DD + d] + ghb[DD + d])));
  float n = tanhf(gib[2 * DD + d] + r * ghb[2 * DD + d]);
  float dv = (1.0f - z) * n + z * deter_f[idx];
  deter_f[idx] = dv;
  u16 bv = f2bf(dv);
  deter_bf[idx] = bv;
  post_in[(size_t)b * 2048 + d] = bv;
  out_deter[((size_t)b * TT + t) * DD + d] = dv;
}

// posterior sample + output scatter
__global__ void step_finish(const float* __restrict__ pl, const float* __restrict__ ql,
                            const float* __restrict__ noise, float* __restrict__ stoch_f,
                            float* __restrict__ oStoch, float* __restrict__ oPm,
                            float* __restrict__ oPs, float* __restrict__ oQm,
                            float* __restrict__ oQs, int t) {
  int idx = blockIdx.x * blockDim.x + threadIdx.x;   // B*S
  if (idx >= BB * SS) return;
  int b = idx >> 6, s = idx & 63;
  float pm = pl[(size_t)b * 2 * SS + s];
  float ps = softplus_f(pl[(size_t)b * 2 * SS + SS + s]);
  float qm = ql[(size_t)b * 2 * SS + s];
  float qs = softplus_f(ql[(size_t)b * 2 * SS + SS + s]);
  float eps = noise[((size_t)t * BB + b) * SS + s];
  float st = qm + qs * eps;
  stoch_f[idx] = st;
  size_t o = ((size_t)b * TT + t) * SS + s;
  oStoch[o] = st; oPm[o] = pm; oPs[o] = ps;
  if (t == TT - 1) { oQm[(size_t)b * SS + s] = qm; oQs[(size_t)b * SS + s] = qs; }
}

// -------------------------------- launcher ----------------------------------
extern "C" void kernel_launch(void* const* d_in, const int* in_sizes, int n_in,
                              void* d_out, int out_size, void* d_ws, size_t ws_size,
                              hipStream_t stream) {
  const float* emb      = (const float*)d_in[0];
  const float* acts     = (const float*)d_in[1];
  const int*   isf      = (const int*)  d_in[2];
  const float* noise    = (const float*)d_in[3];
  const float* w_as     = (const float*)d_in[4];
  const float* b_as     = (const float*)d_in[5];
  const float* w_ih     = (const float*)d_in[6];
  const float* w_hh     = (const float*)d_in[7];
  const float* b_ih     = (const float*)d_in[8];
  const float* b_hh     = (const float*)d_in[9];
  const float* post_w1  = (const float*)d_in[10];
  const float* post_b1  = (const float*)d_in[11];
  const float* post_w2  = (const float*)d_in[12];
  const float* post_b2  = (const float*)d_in[13];
  const float* prior_w1 = (const float*)d_in[14];
  const float* prior_b1 = (const float*)d_in[15];
  const float* prior_w2 = (const float*)d_in[16];
  const float* prior_b2 = (const float*)d_in[17];
  float* out = (float*)d_out;

  // output layout (reference return order, flattened)
  const size_t OFF_STOCH = 0;
  const size_t OFF_DETER = (size_t)BB * TT * SS;
  const size_t OFF_PM    = OFF_DETER + (size_t)BB * TT * DD;
  const size_t OFF_PS    = OFF_PM + (size_t)BB * TT * SS;
  const size_t OFF_QM    = OFF_PS + (size_t)BB * TT * SS;
  const size_t OFF_QS    = OFF_QM + (size_t)BB * SS;

  // workspace bump allocator (256B aligned)
  char* ws = (char*)d_ws;
  size_t off = 0;
  auto alloc = [&](size_t bytes) -> void* {
    void* p = ws + off;
    off += (bytes + 255) & ~(size_t)255;
    return p;
  };

  u16* w_as_c  = (u16*)alloc((size_t)HH * KAS * 2);
  u16* w_ih_c  = (u16*)alloc((size_t)3 * DD * HH * 2);
  u16* w_hh_c  = (u16*)alloc((size_t)3 * DD * DD * 2);
  u16* pr_w1_c = (u16*)alloc((size_t)HH * DD * 2);
  u16* pr_w2_c = (u16*)alloc((size_t)2 * SS * HH * 2);
  u16* po_w1_c = (u16*)alloc((size_t)HH * 2048 * 2);
  u16* po_w2_c = (u16*)alloc((size_t)2 * SS * HH * 2);

  u16* x_as    = (u16*)alloc((size_t)BB * KAS * 2);
  u16* x_bf    = (u16*)alloc((size_t)BB * HH * 2);
  u16* deterbf = (u16*)alloc((size_t)BB * DD * 2);
  u16* ph      = (u16*)alloc((size_t)BB * HH * 2);
  u16* qh      = (u16*)alloc((size_t)BB * HH * 2);
  u16* post_in = (u16*)alloc((size_t)BB * 2048 * 2);

  float* stoch_f = (float*)alloc((size_t)BB * SS * 4);
  float* deter_f = (float*)alloc((size_t)BB * DD * 4);
  float* gi      = (float*)alloc((size_t)BB * 3 * DD * 4);
  float* gh      = (float*)alloc((size_t)BB * 3 * DD * 4);
  float* pl      = (float*)alloc((size_t)BB * 2 * SS * 4);
  float* ql      = (float*)alloc((size_t)BB * 2 * SS * 4);

  auto cvt = [&](const float* s, u16* d, int N, int K, int Kpad) {
    int tot = N * Kpad;
    convert_pad_bf16<<<(tot + 255) / 256, 256, 0, stream>>>(s, d, N, K, Kpad);
  };
  cvt(w_as,     w_as_c,  HH,     AA + SS, KAS);
  cvt(w_ih,     w_ih_c,  3 * DD, HH,      HH);
  cvt(w_hh,     w_hh_c,  3 * DD, DD,      DD);
  cvt(prior_w1, pr_w1_c, HH,     DD,      DD);
  cvt(prior_w2, pr_w2_c, 2 * SS, HH,      HH);
  cvt(post_w1,  po_w1_c, HH,     2048,    2048);
  cvt(post_w2,  po_w2_c, 2 * SS, HH,      HH);

  zero_f32<<<(BB * SS + 255) / 256, 256, 0, stream>>>(stoch_f, BB * SS);
  zero_f32<<<(BB * DD + 255) / 256, 256, 0, stream>>>(deter_f, BB * DD);

  auto gemm = [&](const u16* A, int lda, const u16* W, int ldw, const float* bias,
                  void* C, int N, int K, int flags) {
    dim3 g(BB / 128, N / 128), b(256);
    wmma_gemm<<<g, b, 0, stream>>>(A, lda, W, ldw, bias, C, N, K, flags);
  };

  const int eDim = BB * DD, eS = BB * SS;
  for (int t = 0; t < TT; ++t) {
    step_prep<<<(eDim + 255) / 256, 256, 0, stream>>>(
        acts, emb, isf, stoch_f, deter_f, deterbf, x_as, post_in, t);

    // x = [a|stoch]·w_asᵀ + b_as              (out bf16, no act)
    gemm(x_as, KAS, w_as_c, KAS, b_as, x_bf, HH, KAS, /*flags=*/2);
    // gi = x·w_ihᵀ + b_ih ; gh = deter·w_hhᵀ + b_hh   (fp32)
    gemm(x_bf,    HH, w_ih_c, HH, b_ih, gi, 3 * DD, HH, 0);
    gemm(deterbf, DD, w_hh_c, DD, b_hh, gh, 3 * DD, DD, 0);

    gru_gate<<<(eDim + 255) / 256, 256, 0, stream>>>(
        gi, gh, deter_f, deterbf, post_in, out + OFF_DETER, t);

    // prior MLP: ELU layer1 (bf16) then linear layer2 (fp32)
    gemm(deterbf, DD, pr_w1_c, DD, prior_b1, ph, HH, DD, 1 | 2);
    gemm(ph,      HH, pr_w2_c, HH, prior_b2, pl, 2 * SS, HH, 0);
    // post MLP on [deter|e] (K=2048)
    gemm(post_in, 2048, po_w1_c, 2048, post_b1, qh, HH, 2048, 1 | 2);
    gemm(qh,      HH,   po_w2_c, HH,   post_b2, ql, 2 * SS, HH, 0);

    step_finish<<<(eS + 255) / 256, 256, 0, stream>>>(
        pl, ql, noise, stoch_f,
        out + OFF_STOCH, out + OFF_PM, out + OFF_PS, out + OFF_QM, out + OFF_QS, t);
  }
  (void)in_sizes; (void)n_in; (void)out_size; (void)ws_size;
}
